// UnifiedDeepRanker_42511586295930
// MI455X (gfx1250) — matrix operand order repaired and checked by reading, not compiled, not run on hardware
//
#include <hip/hip_runtime.h>
#include <hip/hip_bf16.h>
#include <math.h>

#define BATCH 8192
#define LSEQ  50
#define DEMB  128
#define DIN_D 392
#define DPAD  416
#define NEXP  4
#define NTASK 2
#define H1    256
#define H2    128
#define ATTH  64
#define NCROSS 3
#define NPADC 448

typedef __attribute__((ext_vector_type(16))) __bf16 v16bf;
typedef __attribute__((ext_vector_type(8)))  float  v8f;

union AFrag { v16bf v; unsigned u[8]; };

// Native conversions: let the backend emit v_cvt_pk_bf16_f32 instead of
// software RNE bit-twiddling (prev round showed v_add3/v_and/v_mov_b16 bloat).
__device__ inline __bf16 to_bf16(float f) { return (__bf16)f; }
__device__ inline unsigned pack2_bf16(float lo, float hi) {
  union { __bf16 b[2]; unsigned u; } o;
  o.b[0] = (__bf16)lo;
  o.b[1] = (__bf16)hi;
  return o.u;
}

// ---------------------------------------------------------------------------
// Shared 64x64 output-tile WMMA core: A row-major [M][lda] bf16, BT = B^T
// stored K-major per output column: BT[n][ldb]. Per-wave 16(M) x 64(N) strip.
// A-fragment pair layout (16-bit A 16x32): VGPR j holds K = (j<4?2j:2j+8)+8*half
// B-fragment (32x16): VGPR j holds K = (j<4?2j:2j+8)+8*half, N = lane&15.
// C/D: N = lane&15, M = 8*half + r.
// ---------------------------------------------------------------------------
__device__ inline void wmma_tile_64x64(const __bf16* __restrict__ A, int lda,
                                       const __bf16* __restrict__ BT, int ldb,
                                       int ksteps, int mbase, int nbase,
                                       v8f acc[4]) {
  const int lane = threadIdx.x & 31;
  const int half = lane >> 4;
  const int nlo  = lane & 15;
  const __bf16* Arow = A + (size_t)(mbase + nlo) * lda;
  for (int ks = 0; ks < ksteps; ++ks) {
    const int k0 = ks * 32;
    AFrag a;
#pragma unroll
    for (int j = 0; j < 8; ++j) {
      const int kp = ((j < 4) ? 2 * j : 8 + 2 * j) + half * 8;
      a.u[j] = *(const unsigned*)(Arow + k0 + kp);
    }
#pragma unroll
    for (int nt = 0; nt < 4; ++nt) {
      const int n = nbase + nt * 16 + nlo;
      const __bf16* Brow = BT + (size_t)n * ldb + k0 + half * 8;
      AFrag bfr;
#pragma unroll
      for (int j = 0; j < 8; ++j) bfr.u[j] = *(const unsigned*)(Brow + ((j < 4) ? 2 * j : 8 + 2 * j));
      acc[nt] = __builtin_amdgcn_wmma_f32_16x16x32_bf16(
          false, a.v, false, bfr.v, (short)0, acc[nt], false, false);
    }
  }
}

// ---------------------------------------------------------------------------
// K0: pack weights to bf16, transposed K-major, zero-padded
// ---------------------------------------------------------------------------
__global__ void prep_weights_kernel(const float* __restrict__ w1,
                                    const float* __restrict__ crossW,
                                    const float* __restrict__ expW1,
                                    const float* __restrict__ expW2,
                                    __bf16* __restrict__ WcT,
                                    __bf16* __restrict__ crossWT,
                                    __bf16* __restrict__ expW1T,
                                    __bf16* __restrict__ expW2T) {
  const int NWc = ATTH * 256;
  const int NCr = NCROSS * NPADC * DPAD;
  const int NE1 = NEXP * H1 * DPAD;
  const int NE2 = NEXP * H2 * H1;
  const int total = NWc + NCr + NE1 + NE2;
  for (int i = blockIdx.x * blockDim.x + threadIdx.x; i < total;
       i += gridDim.x * blockDim.x) {
    if (i < NWc) {
      // combined DIN weight: rows 0..127 = Wk - Wd, rows 128..255 = Wp
      int n = i / 256, k = i % 256;
      float v;
      if (k < 128) v = w1[(128 + k) * ATTH + n] - w1[(256 + k) * ATTH + n];
      else         v = w1[(384 + (k - 128)) * ATTH + n];
      WcT[n * 256 + k] = to_bf16(v);
    } else if (i < NWc + NCr) {
      int j = i - NWc;
      int l = j / (NPADC * DPAD);
      int rem = j % (NPADC * DPAD);
      int n = rem / DPAD, k = rem % DPAD;
      float v = (n < DIN_D && k < DIN_D)
                    ? crossW[(size_t)l * DIN_D * DIN_D + (size_t)k * DIN_D + n]
                    : 0.f;
      crossWT[j] = to_bf16(v);
    } else if (i < NWc + NCr + NE1) {
      int j = i - NWc - NCr;
      int e = j / (H1 * DPAD);
      int rem = j % (H1 * DPAD);
      int n = rem / DPAD, k = rem % DPAD;
      float v = (k < DIN_D) ? expW1[(size_t)e * DIN_D * H1 + (size_t)k * H1 + n]
                            : 0.f;
      expW1T[j] = to_bf16(v);
    } else {
      int j = i - NWc - NCr - NE1;
      int e = j / (H2 * H1);
      int rem = j % (H2 * H1);
      int n = rem / H1, k = rem % H1;
      expW2T[j] = to_bf16(expW2[(size_t)e * H1 * H2 + (size_t)k * H2 + n]);
    }
  }
}

// ---------------------------------------------------------------------------
// K1: DIN attention + interest pool + concat + input LayerNorm (1 block/row)
// hidden(l,n) = q.(Wq+Wd) + [hist | q*hist](l,:) @ Wc ; relu ; .w2 -> score
// ---------------------------------------------------------------------------
__global__ void __launch_bounds__(128)
din_attention_kernel(const int* __restrict__ userId, const int* __restrict__ movieId,
                     const int* __restrict__ seq, const float* __restrict__ dense,
                     const float* __restrict__ embU, const float* __restrict__ embM,
                     const float* __restrict__ w1, const float* __restrict__ b1,
                     const float* __restrict__ w2, const float* __restrict__ b2,
                     const float* __restrict__ gamma, const float* __restrict__ beta,
                     const __bf16* __restrict__ WcT,
                     float* __restrict__ x0f, __bf16* __restrict__ x0b) {
  const int b = blockIdx.x;
  const int tid = threadIdx.x;
  __shared__ float sQ[DEMB];
  __shared__ float sHist[64 * DEMB];
  __shared__ float sQW[64];
  __shared__ float sW2[64];
  __shared__ float sScore[64];
  __shared__ float sProbs[64];
  __shared__ int   sSeq[64];
  __shared__ float sRed[2];
  __shared__ float rs[128], rq[128];

  const int mid = movieId[b];
  if (tid < 64) {
    sSeq[tid] = (tid < LSEQ) ? seq[b * LSEQ + tid] : 0;
    sW2[tid] = w2[tid];
  }
  sQ[tid] = embM[(size_t)mid * DEMB + tid];
  __syncthreads();
  // gather history rows (pad rows use id 0 -> embedding row 0 == zeros)
  for (int i = tid; i < 64 * DEMB; i += 128) {
    int l = i >> 7, d = i & 127;
    sHist[i] = embM[(size_t)sSeq[l] * DEMB + d];
  }
  // per-row bias: q.(Wq+Wd) + b1 ; init scores with b2
  if (tid < 64) {
    float s = b1[tid];
    for (int k = 0; k < DEMB; ++k)
      s += sQ[k] * (w1[k * ATTH + tid] + w1[(256 + k) * ATTH + tid]);
    sQW[tid] = s;
    sScore[tid] = b2[0];
  }
  __syncthreads();

  // WMMA: A = [hist | q*hist] (64 x 256), B = Wc (256 x 64)
  {
    const int wave = tid >> 5, lane = tid & 31;
    const int half = lane >> 4, nlo = lane & 15;
    const int mbase = wave * 16;
    const int mA = mbase + nlo;
    v8f acc[4] = {};
    for (int ks = 0; ks < 8; ++ks) {
      const int k0 = ks * 32;
      AFrag a;
#pragma unroll
      for (int j = 0; j < 8; ++j) {
        const int kp = ((j < 4) ? 2 * j : 8 + 2 * j) + half * 8;
        const int kg = k0 + kp;
        float v0, v1;
        if (kg < DEMB) {
          v0 = sHist[mA * DEMB + kg];
          v1 = sHist[mA * DEMB + kg + 1];
        } else {
          const int kk = kg - DEMB;
          v0 = sHist[mA * DEMB + kk] * sQ[kk];
          v1 = sHist[mA * DEMB + kk + 1] * sQ[kk + 1];
        }
        a.u[j] = pack2_bf16(v0, v1);
      }
#pragma unroll
      for (int nt = 0; nt < 4; ++nt) {
        const int n = nt * 16 + nlo;
        const __bf16* Brow = WcT + (size_t)n * 256 + k0 + half * 8;
        AFrag bfr;
#pragma unroll
        for (int j = 0; j < 8; ++j) bfr.u[j] = *(const unsigned*)(Brow + ((j < 4) ? 2 * j : 8 + 2 * j));
        acc[nt] = __builtin_amdgcn_wmma_f32_16x16x32_bf16(
            false, a.v, false, bfr.v, (short)0, acc[nt], false, false);
      }
    }
    const int mh = half * 8;
#pragma unroll
    for (int nt = 0; nt < 4; ++nt) {
      const int n = nt * 16 + nlo;
#pragma unroll
      for (int r = 0; r < 8; ++r) {
        const int m = mbase + mh + r;
        float h = fmaxf(acc[nt][r] + sQW[n], 0.f);
        atomicAdd(&sScore[m], h * sW2[n]);
      }
    }
  }
  __syncthreads();
  // mask + softmax over L
  if (tid < 64) {
    float sc = (tid < LSEQ && sSeq[tid] > 0) ? sScore[tid] : -1e9f;
    sScore[tid] = sc;
  }
  __syncthreads();
  if (tid == 0) {
    float mx = -1e30f;
    for (int l = 0; l < LSEQ; ++l) mx = fmaxf(mx, sScore[l]);
    sRed[0] = mx;
  }
  __syncthreads();
  if (tid < 64) sProbs[tid] = (tid < LSEQ) ? __expf(sScore[tid] - sRed[0]) : 0.f;
  __syncthreads();
  if (tid == 0) {
    float s = 0.f;
    for (int l = 0; l < LSEQ; ++l) s += sProbs[l];
    sRed[1] = 1.f / s;
  }
  __syncthreads();
  // interest pooling + concat + LayerNorm
  const float pinv = sRed[1];
  float xi = 0.f;
  for (int l = 0; l < LSEQ; ++l) xi += sProbs[l] * pinv * sHist[l * DEMB + tid];
  const int uid = userId[b];
  const float xu = embU[(size_t)uid * DEMB + tid];
  const float xm = sQ[tid];
  const float xd = (tid < 8) ? dense[b * 8 + tid] : 0.f;
  rs[tid] = xu + xm + xi + xd;
  rq[tid] = xu * xu + xm * xm + xi * xi + xd * xd;
  __syncthreads();
  for (int s = 64; s > 0; s >>= 1) {
    if (tid < s) { rs[tid] += rs[tid + s]; rq[tid] += rq[tid + s]; }
    __syncthreads();
  }
  const float mean = rs[0] / (float)DIN_D;
  const float var  = rq[0] / (float)DIN_D - mean * mean;
  const float rinv = rsqrtf(var + 1e-5f);
  float*  xrow  = x0f + (size_t)b * DIN_D;
  __bf16* xrowb = x0b + (size_t)b * DPAD;
  float v;
  v = (xu - mean) * rinv * gamma[tid] + beta[tid];
  xrow[tid] = v; xrowb[tid] = to_bf16(v);
  v = (xm - mean) * rinv * gamma[DEMB + tid] + beta[DEMB + tid];
  xrow[DEMB + tid] = v; xrowb[DEMB + tid] = to_bf16(v);
  v = (xi - mean) * rinv * gamma[256 + tid] + beta[256 + tid];
  xrow[256 + tid] = v; xrowb[256 + tid] = to_bf16(v);
  if (tid < 8) {
    v = (xd - mean) * rinv * gamma[384 + tid] + beta[384 + tid];
    xrow[384 + tid] = v; xrowb[384 + tid] = to_bf16(v);
  }
  if (tid < DPAD - DIN_D) xrowb[DIN_D + tid] = to_bf16(0.f);
}

// ---------------------------------------------------------------------------
// K2: one CrossNetV2 layer: out = x0 * (x @ W + b) + x
// ---------------------------------------------------------------------------
__global__ void __launch_bounds__(128)
cross_kernel(const float* __restrict__ xin, const __bf16* __restrict__ xinb,
             const float* __restrict__ x0, const __bf16* __restrict__ WT,
             const float* __restrict__ bias,
             float* __restrict__ xout, __bf16* __restrict__ xoutb) {
  const int wave = threadIdx.x >> 5, lane = threadIdx.x & 31;
  const int mbase = blockIdx.x * 64 + wave * 16;
  const int nbase = blockIdx.y * 64;
  v8f acc[4] = {};
  wmma_tile_64x64(xinb, DPAD, WT, DPAD, DPAD / 32, mbase, nbase, acc);
  const int nlo = lane & 15, mh = (lane >> 4) * 8;
#pragma unroll
  for (int nt = 0; nt < 4; ++nt) {
    const int n = nbase + nt * 16 + nlo;
#pragma unroll
    for (int r = 0; r < 8; ++r) {
      const int m = mbase + mh + r;
      if (n < DIN_D) {
        float xw = acc[nt][r] + bias[n];
        float v = x0[(size_t)m * DIN_D + n] * xw + xin[(size_t)m * DIN_D + n];
        xout[(size_t)m * DIN_D + n] = v;
        xoutb[(size_t)m * DPAD + n] = to_bf16(v);
      } else if (n < DPAD) {
        xoutb[(size_t)m * DPAD + n] = to_bf16(0.f);
      }
    }
  }
}

// ---------------------------------------------------------------------------
// K3: cross LayerNorm + MMoE gate softmax
// ---------------------------------------------------------------------------
__global__ void __launch_bounds__(128)
ln_gate_kernel(const float* __restrict__ xin, const float* __restrict__ gamma,
               const float* __restrict__ beta, const float* __restrict__ gateW,
               const float* __restrict__ gateB,
               __bf16* __restrict__ xnb, float* __restrict__ g) {
  const int b = blockIdx.x, tid = threadIdx.x;
  __shared__ float sx[DIN_D];
  __shared__ float rs[128], rq[128];
  __shared__ float sg[8];
  float ls = 0.f, lq = 0.f;
  for (int i = tid; i < DIN_D; i += 128) {
    float v = xin[(size_t)b * DIN_D + i];
    sx[i] = v; ls += v; lq += v * v;
  }
  rs[tid] = ls; rq[tid] = lq;
  __syncthreads();
  for (int s = 64; s > 0; s >>= 1) {
    if (tid < s) { rs[tid] += rs[tid + s]; rq[tid] += rq[tid + s]; }
    __syncthreads();
  }
  const float mean = rs[0] / (float)DIN_D;
  const float rinv = rsqrtf(rq[0] / (float)DIN_D - mean * mean + 1e-5f);
  for (int i = tid; i < DIN_D; i += 128) {
    float v = (sx[i] - mean) * rinv * gamma[i] + beta[i];
    sx[i] = v;
    xnb[(size_t)b * DPAD + i] = to_bf16(v);
  }
  if (tid < DPAD - DIN_D) xnb[(size_t)b * DPAD + DIN_D + tid] = to_bf16(0.f);
  __syncthreads();
  if (tid < NTASK * NEXP) {
    int t = tid >> 2, e = tid & 3;
    float s = gateB[t * NEXP + e];
    for (int k = 0; k < DIN_D; ++k) s += sx[k] * gateW[t * DIN_D * NEXP + k * NEXP + e];
    sg[tid] = s;
  }
  __syncthreads();
  if (tid < NTASK) {
    const float* l = sg + tid * 4;
    float mx = fmaxf(fmaxf(l[0], l[1]), fmaxf(l[2], l[3]));
    float e0 = __expf(l[0] - mx), e1 = __expf(l[1] - mx);
    float e2 = __expf(l[2] - mx), e3 = __expf(l[3] - mx);
    float inv = 1.f / (e0 + e1 + e2 + e3);
    float* gr = g + ((size_t)tid * BATCH + b) * NEXP;
    gr[0] = e0 * inv; gr[1] = e1 * inv; gr[2] = e2 * inv; gr[3] = e3 * inv;
  }
}

// ---------------------------------------------------------------------------
// K4: expert layer 1 (relu) -> h bf16 ; K5: expert layer 2 -> eo f32
// ---------------------------------------------------------------------------
__global__ void __launch_bounds__(128)
expert1_kernel(const __bf16* __restrict__ xnb, const __bf16* __restrict__ W1T,
               const float* __restrict__ b1, __bf16* __restrict__ h) {
  const int e = blockIdx.z;
  const int wave = threadIdx.x >> 5, lane = threadIdx.x & 31;
  const int mbase = blockIdx.x * 64 + wave * 16;
  const int nbase = blockIdx.y * 64;
  v8f acc[4] = {};
  wmma_tile_64x64(xnb, DPAD, W1T + (size_t)e * H1 * DPAD, DPAD, DPAD / 32,
                  mbase, nbase, acc);
  const int nlo = lane & 15, mh = (lane >> 4) * 8;
  __bf16* he = h + (size_t)e * BATCH * H1;
#pragma unroll
  for (int nt = 0; nt < 4; ++nt) {
    const int n = nbase + nt * 16 + nlo;
    const float bias = b1[e * H1 + n];
#pragma unroll
    for (int r = 0; r < 8; ++r) {
      const int m = mbase + mh + r;
      he[(size_t)m * H1 + n] = to_bf16(fmaxf(acc[nt][r] + bias, 0.f));
    }
  }
}

__global__ void __launch_bounds__(128)
expert2_kernel(const __bf16* __restrict__ h, const __bf16* __restrict__ W2T,
               const float* __restrict__ b2, float* __restrict__ eo) {
  const int e = blockIdx.z;
  const int wave = threadIdx.x >> 5, lane = threadIdx.x & 31;
  const int mbase = blockIdx.x * 64 + wave * 16;
  const int nbase = blockIdx.y * 64;
  v8f acc[4] = {};
  wmma_tile_64x64(h + (size_t)e * BATCH * H1, H1, W2T + (size_t)e * H2 * H1, H1,
                  H1 / 32, mbase, nbase, acc);
  const int nlo = lane & 15, mh = (lane >> 4) * 8;
  float* eoe = eo + (size_t)e * BATCH * H2;
#pragma unroll
  for (int nt = 0; nt < 4; ++nt) {
    const int n = nbase + nt * 16 + nlo;
    const float bias = b2[e * H2 + n];
#pragma unroll
    for (int r = 0; r < 8; ++r) {
      const int m = mbase + mh + r;
      eoe[(size_t)m * H2 + n] = acc[nt][r] + bias;
    }
  }
}

// ---------------------------------------------------------------------------
// K6: gate-weighted expert mix + task heads -> logits [T, B]
// ---------------------------------------------------------------------------
__global__ void __launch_bounds__(128)
combine_head_kernel(const float* __restrict__ g, const float* __restrict__ eo,
                    const float* __restrict__ headW, const float* __restrict__ headB,
                    float* __restrict__ out) {
  const int b = blockIdx.x, tid = threadIdx.x;
  __shared__ float rbuf[128];
  for (int t = 0; t < NTASK; ++t) {
    const float* gr = g + ((size_t)t * BATCH + b) * NEXP;
    float s = 0.f;
#pragma unroll
    for (int e = 0; e < NEXP; ++e)
      s += gr[e] * eo[((size_t)e * BATCH + b) * H2 + tid];
    rbuf[tid] = s * headW[t * H2 + tid];
    __syncthreads();
    for (int st = 64; st > 0; st >>= 1) {
      if (tid < st) rbuf[tid] += rbuf[tid + st];
      __syncthreads();
    }
    if (tid == 0) out[(size_t)t * BATCH + b] = rbuf[0] + headB[t];
    __syncthreads();
  }
}

// ---------------------------------------------------------------------------
extern "C" void kernel_launch(void* const* d_in, const int* in_sizes, int n_in,
                              void* d_out, int out_size, void* d_ws, size_t ws_size,
                              hipStream_t stream) {
  (void)in_sizes; (void)n_in; (void)out_size; (void)ws_size;
  const int*   userId  = (const int*)d_in[0];
  const int*   movieId = (const int*)d_in[1];
  const int*   seq     = (const int*)d_in[2];
  const float* dense   = (const float*)d_in[3];
  const float* embU    = (const float*)d_in[4];
  const float* embM    = (const float*)d_in[5];
  const float* att_w1  = (const float*)d_in[6];
  const float* att_b1  = (const float*)d_in[7];
  const float* att_w2  = (const float*)d_in[8];
  const float* att_b2  = (const float*)d_in[9];
  const float* in_g    = (const float*)d_in[10];
  const float* in_b    = (const float*)d_in[11];
  const float* cr_g    = (const float*)d_in[12];
  const float* cr_b    = (const float*)d_in[13];
  const float* cross_W = (const float*)d_in[14];
  const float* cross_b = (const float*)d_in[15];
  const float* exp_w1  = (const float*)d_in[16];
  const float* exp_b1  = (const float*)d_in[17];
  const float* exp_w2  = (const float*)d_in[18];
  const float* exp_b2  = (const float*)d_in[19];
  const float* gate_w  = (const float*)d_in[20];
  const float* gate_b  = (const float*)d_in[21];
  const float* head_w  = (const float*)d_in[22];
  const float* head_b  = (const float*)d_in[23];
  float* out = (float*)d_out;

  char* ws = (char*)d_ws;
  size_t off = 0;
  auto take = [&](size_t bytes) {
    size_t o = off;
    off += (bytes + 255) & ~(size_t)255;
    return o;
  };
  __bf16* WcT     = (__bf16*)(ws + take((size_t)ATTH * 256 * 2));
  __bf16* crossWT = (__bf16*)(ws + take((size_t)NCROSS * NPADC * DPAD * 2));
  __bf16* expW1T  = (__bf16*)(ws + take((size_t)NEXP * H1 * DPAD * 2));
  __bf16* expW2T  = (__bf16*)(ws + take((size_t)NEXP * H2 * H1 * 2));
  float*  x0f     = (float*) (ws + take((size_t)BATCH * DIN_D * 4));
  __bf16* x0b     = (__bf16*)(ws + take((size_t)BATCH * DPAD * 2));
  float*  xAf     = (float*) (ws + take((size_t)BATCH * DIN_D * 4));
  __bf16* xAb     = (__bf16*)(ws + take((size_t)BATCH * DPAD * 2));
  float*  xBf     = (float*) (ws + take((size_t)BATCH * DIN_D * 4));
  __bf16* xBb     = (__bf16*)(ws + take((size_t)BATCH * DPAD * 2));
  __bf16* xnb     = (__bf16*)(ws + take((size_t)BATCH * DPAD * 2));
  __bf16* hb      = (__bf16*)(ws + take((size_t)NEXP * BATCH * H1 * 2));
  float*  eo      = (float*) (ws + take((size_t)NEXP * BATCH * H2 * 4));
  float*  g       = (float*) (ws + take((size_t)NTASK * BATCH * NEXP * 4));

  prep_weights_kernel<<<1024, 256, 0, stream>>>(att_w1, cross_W, exp_w1, exp_w2,
                                                WcT, crossWT, expW1T, expW2T);
  din_attention_kernel<<<BATCH, 128, 0, stream>>>(
      userId, movieId, seq, dense, embU, embM, att_w1, att_b1, att_w2, att_b2,
      in_g, in_b, WcT, x0f, x0b);
  dim3 cgrid(BATCH / 64, NPADC / 64, 1);
  cross_kernel<<<cgrid, 128, 0, stream>>>(
      x0f, x0b, x0f, crossWT + 0 * (size_t)NPADC * DPAD, cross_b + 0 * DIN_D, xAf, xAb);
  cross_kernel<<<cgrid, 128, 0, stream>>>(
      xAf, xAb, x0f, crossWT + 1 * (size_t)NPADC * DPAD, cross_b + 1 * DIN_D, xBf, xBb);
  cross_kernel<<<cgrid, 128, 0, stream>>>(
      xBf, xBb, x0f, crossWT + 2 * (size_t)NPADC * DPAD, cross_b + 2 * DIN_D, xAf, xAb);
  ln_gate_kernel<<<BATCH, 128, 0, stream>>>(xAf, cr_g, cr_b, gate_w, gate_b, xnb, g);
  expert1_kernel<<<dim3(BATCH / 64, H1 / 64, NEXP), 128, 0, stream>>>(xnb, expW1T, exp_b1, hb);
  expert2_kernel<<<dim3(BATCH / 64, H2 / 64, NEXP), 128, 0, stream>>>(hb, expW2T, exp_b2, eo);
  combine_head_kernel<<<BATCH, 128, 0, stream>>>(g, eo, head_w, head_b, out);
}
